// MTPAttention_28819230556765
// MI455X (gfx1250) — matrix operand order repaired and checked by reading, not compiled, not run on hardware
//
#include <hip/hip_runtime.h>

// ---------------------------------------------------------------------------
// Types for CDNA5 WMMA (wave32): v_wmma_f32_16x16x32_bf16
// ---------------------------------------------------------------------------
typedef __bf16 bf16_t;
typedef __attribute__((ext_vector_type(16))) __bf16 v16bf;
typedef __attribute__((ext_vector_type(8)))  float  v8f;

#define WMMA_BF16(a, b, c) \
  __builtin_amdgcn_wmma_f32_16x16x32_bf16(false, (a), false, (b), (short)0, (c), false, false)

// Problem constants
#define B_   2
#define S_   2048
#define HS_  1024
#define H_   16
#define KV_  4
#define G_   4
#define DH_  64
#define MTOT (B_ * S_)          // 4096 rows
#define SCALE_ 0.125f           // 64^-0.5
#define EPS_   1e-6f

// ---------------------------------------------------------------------------
// CDNA5 async global->LDS copy (ASYNCcnt path) + wait
// Builtin signature (probe-confirmed by diagnostic): param1 = AS(1) int4*,
// param2 = AS(3) int4*, then imm offset, imm cpol.
// ---------------------------------------------------------------------------
typedef int v4i_ __attribute__((vector_size(16)));
typedef __attribute__((address_space(1))) v4i_* g_v4i;
typedef __attribute__((address_space(3))) v4i_* l_v4i;

__device__ __forceinline__ void async_b128(const void* g, void* l) {
#if __has_builtin(__builtin_amdgcn_global_load_async_to_lds_b128)
  __builtin_amdgcn_global_load_async_to_lds_b128((g_v4i)g, (l_v4i)l, 0, 0);
#else
  unsigned loff = (unsigned)(unsigned long long)l;   // low 32 bits = LDS offset
  asm volatile("global_load_async_to_lds_b128 %0, %1, off"
               :: "v"(loff), "v"(g) : "memory");
#endif
}

__device__ __forceinline__ void wait_async0() {
#if __has_builtin(__builtin_amdgcn_s_wait_asynccnt)
  __builtin_amdgcn_s_wait_asynccnt(0);
#else
  asm volatile("s_wait_asynccnt 0x0" ::: "memory");
#endif
}

__device__ __forceinline__ bf16_t f2bf(float x) {
  union { float f; unsigned u; } v; v.f = x;
  unsigned r = (v.u + 0x7FFFu + ((v.u >> 16) & 1u)) >> 16;   // RNE
  union { unsigned short s; bf16_t b; } o; o.s = (unsigned short)r;
  return o.b;
}

// A-fragment (16x32 bf16): lane L holds row (L&15), K runs [kb,kb+8) and
// [kb+16,kb+24) with kb = 8*(L>>4).
__device__ __forceinline__ v16bf frag_a(const bf16_t* base, int ld, int row, int lane) {
  const bf16_t* p = base + (row + (lane & 15)) * ld + ((lane >> 4) << 3);
  v16bf f;
  ((float4*)&f)[0] = *(const float4*)(p);
  ((float4*)&f)[1] = *(const float4*)(p + 16);
  return f;
}

// B-fragment (32x16 bf16): lane L holds col (L&15), K runs [kb,kb+8) and
// [kb+16,kb+24) with kb = 8*(L>>4).
// base is a [N][K] (K-contiguous) LDS tile.
__device__ __forceinline__ v16bf frag_b(const bf16_t* base, int ld, int col, int lane) {
  const bf16_t* p = base + (col + (lane & 15)) * ld + ((lane >> 4) << 3);
  v16bf f;
  ((float4*)&f)[0] = *(const float4*)(p);
  ((float4*)&f)[1] = *(const float4*)(p + 16);
  return f;
}

// ---------------------------------------------------------------------------
// fp32 -> bf16 bulk conversion (n multiple of 1024)
// ---------------------------------------------------------------------------
__global__ __launch_bounds__(256) void cvt_bf16_kernel(const float* __restrict__ in,
                                                       bf16_t* __restrict__ out, int n) {
  int i = (blockIdx.x * 256 + threadIdx.x) * 4;
  if (i >= n) return;
  float4 v = *(const float4*)(in + i);
  bf16_t t[4] = { f2bf(v.x), f2bf(v.y), f2bf(v.z), f2bf(v.w) };
  *(uint2*)(out + i) = *(uint2*)t;
}

// fp32 [K][N] -> bf16 [N][K] transpose-convert (32x32 LDS tiles)
__global__ __launch_bounds__(256)
void cvt_t_bf16_kernel(const float* __restrict__ in, bf16_t* __restrict__ out,
                       int K, int N) {
  __shared__ float t[32][33];
  const int n0 = blockIdx.x * 32, k0 = blockIdx.y * 32;
  const int tx = threadIdx.x & 31, ty = threadIdx.x >> 5;   // ty: 0..7
#pragma unroll
  for (int i = ty; i < 32; i += 8) t[i][tx] = in[(size_t)(k0 + i) * N + n0 + tx];
  __syncthreads();
#pragma unroll
  for (int i = ty; i < 32; i += 8)
    out[(size_t)(n0 + i) * K + k0 + tx] = f2bf(t[tx][i]);
}

// ---------------------------------------------------------------------------
// bf16 GEMM with async double-buffered LDS tiles.
//   C[M,N] = A[M,K] * Bt[N,K]^T  (weights pre-transposed, both operands
//   K-contiguous -> pure GLOBAL_LOAD_ASYNC_TO_LDS_B128 tile copies)
// Epilogue modes:
//   normw != 0  : fused per-64-col RMS norm, bf16 out (outB)
//   store_vt==1 : scatter into [b][kv][dh][S] transposed V layout (outB)
//   outF  != 0  : fp32 out
//   else        : plain bf16 out
// Block tile 128x64, K-step 32; 8 waves, each 32x32 (2x2 wmma).
// ---------------------------------------------------------------------------
#define BM 128
#define BN 64
#define BK 32

__device__ __forceinline__ void gemm_issue_tiles(const bf16_t* __restrict__ A,
                                                 const bf16_t* __restrict__ Bt, int K,
                                                 int blockM, int blockN, int k0,
                                                 bf16_t* AsBuf, bf16_t* BsBuf, int tid) {
#pragma unroll
  for (int l = 0; l < 2; ++l) {                 // A: 128x32 = 2 x b128 / thread
    int lin = (tid + l * 256) * 8;
    int r = lin >> 5, c = lin & 31;
    async_b128(A + (size_t)(blockM + r) * K + k0 + c, AsBuf + r * BK + c);
  }
  {                                             // B: 64x32 = 1 x b128 / thread
    int r = tid >> 2, c = (tid & 3) * 8;
    async_b128(Bt + (size_t)(blockN + r) * K + k0 + c, BsBuf + r * BK + c);
  }
}

__global__ __launch_bounds__(256)
void gemm_bf16_kernel(const bf16_t* __restrict__ A, const bf16_t* __restrict__ Bt,
                      float* __restrict__ outF, bf16_t* __restrict__ outB,
                      const float* __restrict__ normw, int store_vt,
                      int M, int N, int K) {
  __shared__ __align__(16) bf16_t As[2][BM * BK];   // 16 KB
  __shared__ __align__(16) bf16_t Bs[2][BN * BK];   //  8 KB  [n][k]
  __shared__ __align__(16) float  Cs[BM * BN];      // 32 KB  (norm staging)

  const int tid  = threadIdx.x;
  const int lane = tid & 31;
  const int wave = tid >> 5;
  const int wm = (wave & 3) * 32;
  const int wn = (wave >> 2) * 32;
  const int blockM = blockIdx.y * BM;
  const int blockN = blockIdx.x * BN;

  v8f acc[2][2] = {};

  gemm_issue_tiles(A, Bt, K, blockM, blockN, 0, As[0], Bs[0], tid);
  wait_async0();
  __syncthreads();

  const int NIT = K / BK;
  for (int it = 0; it < NIT; ++it) {
    const int cur = it & 1;
    if (it + 1 < NIT)   // prefetch next tile under this iteration's WMMAs
      gemm_issue_tiles(A, Bt, K, blockM, blockN, (it + 1) * BK,
                       As[cur ^ 1], Bs[cur ^ 1], tid);

    v16bf af[2], bfr[2];
#pragma unroll
    for (int mt = 0; mt < 2; ++mt) af[mt] = frag_a(As[cur], BK, wm + mt * 16, lane);
#pragma unroll
    for (int nt = 0; nt < 2; ++nt) bfr[nt] = frag_b(Bs[cur], BK, wn + nt * 16, lane);
#pragma unroll
    for (int mt = 0; mt < 2; ++mt)
#pragma unroll
      for (int nt = 0; nt < 2; ++nt)
        acc[mt][nt] = WMMA_BF16(af[mt], bfr[nt], acc[mt][nt]);

    wait_async0();
    __syncthreads();
  }

  const int rhalf = (lane >> 4) << 3;   // 0 or 8
  const int cloc  = lane & 15;

  if (normw) {
#pragma unroll
    for (int mt = 0; mt < 2; ++mt)
#pragma unroll
      for (int nt = 0; nt < 2; ++nt)
#pragma unroll
        for (int i = 0; i < 8; ++i)
          Cs[(wm + mt * 16 + i + rhalf) * BN + wn + nt * 16 + cloc] = acc[mt][nt][i];
    __syncthreads();
    if (tid < BM) {
      float ss = 0.f;
#pragma unroll
      for (int c = 0; c < BN; ++c) { float x = Cs[tid * BN + c]; ss += x * x; }
      float sc = rsqrtf(ss * (1.0f / (float)DH_) + EPS_);
      bf16_t* orow = outB + (size_t)(blockM + tid) * N + blockN;
#pragma unroll
      for (int c = 0; c < BN; ++c) orow[c] = f2bf(Cs[tid * BN + c] * sc * normw[c]);
    }
  } else if (store_vt) {
    // V: scatter into Vt[b][kv][dh][s]  (b = row>>11, s = row&2047)
#pragma unroll
    for (int mt = 0; mt < 2; ++mt)
#pragma unroll
      for (int nt = 0; nt < 2; ++nt)
#pragma unroll
        for (int i = 0; i < 8; ++i) {
          int row = blockM + wm + mt * 16 + i + rhalf;
          int col = blockN + wn + nt * 16 + cloc;
          int b = row >> 11, s = row & (S_ - 1);
          outB[((size_t)((b * KV_ + (col >> 6)) * DH_ + (col & 63))) * S_ + s] =
              f2bf(acc[mt][nt][i]);
        }
  } else if (outF) {
#pragma unroll
    for (int mt = 0; mt < 2; ++mt)
#pragma unroll
      for (int nt = 0; nt < 2; ++nt)
#pragma unroll
        for (int i = 0; i < 8; ++i)
          outF[(size_t)(blockM + wm + mt * 16 + i + rhalf) * N + blockN + wn + nt * 16 + cloc] =
              acc[mt][nt][i];
  } else {
#pragma unroll
    for (int mt = 0; mt < 2; ++mt)
#pragma unroll
      for (int nt = 0; nt < 2; ++nt)
#pragma unroll
        for (int i = 0; i < 8; ++i)
          outB[(size_t)(blockM + wm + mt * 16 + i + rhalf) * N + blockN + wn + nt * 16 + cloc] =
              f2bf(acc[mt][nt][i]);
  }
}

// ---------------------------------------------------------------------------
// Flash attention per (b, head, 64-row q-tile). 256 threads, 8 waves.
// Q[4096,1024] bf16; K[4096,256] bf16; Vt[b][kv][64][2048] bf16 (pre-transposed).
// K/V tiles double-buffered via async LDS copies.
// ---------------------------------------------------------------------------
__device__ __forceinline__ void attn_issue_tiles(const bf16_t* __restrict__ Kb,
                                                 const bf16_t* __restrict__ Vt,
                                                 int rowK0, int k0, int kv,
                                                 bf16_t* KsBuf, bf16_t* VtsBuf, int tid) {
#pragma unroll
  for (int l = 0; l < 2; ++l) {
    int lin = (tid + l * 256) * 8;
    int r = lin >> 6, c = lin & 63;
    async_b128(Kb + (size_t)(rowK0 + k0 + r) * (KV_ * DH_) + kv * DH_ + c,
               KsBuf + r * 64 + c);
    async_b128(Vt + (size_t)r * S_ + k0 + c, VtsBuf + r * 64 + c);
  }
}

__global__ __launch_bounds__(256)
void attn_kernel(const bf16_t* __restrict__ Qb, const bf16_t* __restrict__ Kb,
                 const bf16_t* __restrict__ Vt, const float* __restrict__ mask,
                 bf16_t* __restrict__ Ob) {
  __shared__ __align__(16) bf16_t Qs [64 * 64];      //  8 KB [q][dh]
  __shared__ __align__(16) bf16_t Ks [2][64 * 64];   // 16 KB [key][dh]
  __shared__ __align__(16) bf16_t Vts[2][64 * 64];   // 16 KB [dh][key]
  __shared__ __align__(16) float  Ss [64 * 64];      // 16 KB scores fp32
  __shared__ __align__(16) bf16_t Ps [64 * 64];      //  8 KB probs bf16
  __shared__ float corrS[64];
  __shared__ float lS[64];

  const int qt = blockIdx.x, h = blockIdx.y, b = blockIdx.z;
  const int kv = h >> 2;                      // G = 4
  const int tid = threadIdx.x, lane = tid & 31, wave = tid >> 5;
  const int wm = (wave & 3) * 16;
  const int wn = (wave >> 2) * 32;
  const int rhalf = (lane >> 4) << 3;
  const int cloc  = lane & 15;

  const int rowQ0 = b * S_ + qt * 64;
  const int rowK0 = b * S_;
  const bf16_t* Vth = Vt + (size_t)((b * KV_ + kv) * DH_) * S_;

  // async-load Q tile + first K/V tiles
#pragma unroll
  for (int l = 0; l < 2; ++l) {
    int lin = (tid + l * 256) * 8;
    int r = lin >> 6, c = lin & 63;
    async_b128(Qb + (size_t)(rowQ0 + r) * (H_ * DH_) + h * DH_ + c, Qs + r * 64 + c);
  }
  attn_issue_tiles(Kb, Vth, rowK0, 0, kv, Ks[0], Vts[0], tid);
  wait_async0();
  __syncthreads();

  v16bf qa[2];
  qa[0] = frag_a(Qs, 64, wm, lane);           // dh 0..31
  qa[1] = frag_a(Qs + 32, 64, wm, lane);      // dh 32..63

  float m_i = -INFINITY, l_i = 0.f;           // valid on tid<64 (row owner)
  v8f o[2] = {};

  const int NT = S_ / 64;
  for (int kt = 0; kt < NT; ++kt) {
    const int cur = kt & 1;
    if (kt + 1 < NT)    // DMA next K/V tiles under this iteration's math
      attn_issue_tiles(Kb, Vth, rowK0, (kt + 1) * 64, kv, Ks[cur ^ 1], Vts[cur ^ 1], tid);

    // S = Q K^T  (dh contraction, 2 k-steps)
    v8f sc[2] = {};
#pragma unroll
    for (int s = 0; s < 2; ++s)
#pragma unroll
      for (int nt = 0; nt < 2; ++nt) {
        v16bf kf = frag_b(Ks[cur] + s * 32, 64, wn + nt * 16, lane);
        sc[nt] = WMMA_BF16(qa[s], kf, sc[nt]);
      }
#pragma unroll
    for (int nt = 0; nt < 2; ++nt)
#pragma unroll
      for (int i = 0; i < 8; ++i)
        Ss[(wm + i + rhalf) * 64 + wn + nt * 16 + cloc] = sc[nt][i];
    __syncthreads();

    // online softmax: thread t < 64 owns q-row t
    if (tid < 64) {
      const int k0 = kt * 64;
      const float* mrow = mask + (size_t)(qt * 64 + tid) * S_ + k0;
      float mx = m_i;
#pragma unroll 16
      for (int j = 0; j < 64; ++j)
        mx = fmaxf(mx, Ss[tid * 64 + j] * SCALE_ + mrow[j]);
      float corr = __expf(m_i - mx);
      float sum = 0.f;
#pragma unroll 16
      for (int j = 0; j < 64; ++j) {
        float p = __expf(Ss[tid * 64 + j] * SCALE_ + mrow[j] - mx);
        sum += p;
        Ps[tid * 64 + j] = f2bf(p);
      }
      l_i = l_i * corr + sum;
      m_i = mx;
      corrS[tid] = corr;
    }
    __syncthreads();

    // rescale O accumulators, then O += P V  (key contraction, 2 k-steps)
    float cf[8];
#pragma unroll
    for (int i = 0; i < 8; ++i) cf[i] = corrS[wm + i + rhalf];
#pragma unroll
    for (int nt = 0; nt < 2; ++nt)
#pragma unroll
      for (int i = 0; i < 8; ++i) o[nt][i] *= cf[i];

#pragma unroll
    for (int s = 0; s < 2; ++s) {
      v16bf pa = frag_a(Ps + s * 32, 64, wm, lane);
#pragma unroll
      for (int nt = 0; nt < 2; ++nt) {
        v16bf vf = frag_b(Vts[cur] + s * 32, 64, wn + nt * 16, lane);
        o[nt] = WMMA_BF16(pa, vf, o[nt]);
      }
    }
    wait_async0();
    __syncthreads();
  }

  if (tid < 64) lS[tid] = l_i;
  __syncthreads();
  float lf[8];
#pragma unroll
  for (int i = 0; i < 8; ++i) lf[i] = 1.0f / lS[wm + i + rhalf];
#pragma unroll
  for (int nt = 0; nt < 2; ++nt)
#pragma unroll
    for (int i = 0; i < 8; ++i)
      Ob[(size_t)(rowQ0 + wm + i + rhalf) * (H_ * DH_) + h * DH_ + wn + nt * 16 + cloc] =
          f2bf(o[nt][i] * lf[i]);
}

// ---------------------------------------------------------------------------
// Host-side launcher
// ---------------------------------------------------------------------------
extern "C" void kernel_launch(void* const* d_in, const int* in_sizes, int n_in,
                              void* d_out, int out_size, void* d_ws, size_t ws_size,
                              hipStream_t stream) {
  (void)in_sizes; (void)n_in; (void)out_size; (void)ws_size;
  const float* hidden = (const float*)d_in[0];
  const float* mask   = (const float*)d_in[1];
  const float* wq     = (const float*)d_in[2];
  const float* wk     = (const float*)d_in[3];
  const float* wv     = (const float*)d_in[4];
  const float* wo     = (const float*)d_in[5];
  const float* qnw    = (const float*)d_in[6];
  const float* knw    = (const float*)d_in[7];
  float* out = (float*)d_out;

  char* ws = (char*)d_ws;
  size_t off = 0;
  bf16_t* hid_bf = (bf16_t*)(ws + off); off += (size_t)MTOT * HS_ * 2;        // 8 MB
  bf16_t* wqT    = (bf16_t*)(ws + off); off += (size_t)HS_ * (H_ * DH_) * 2;  // 2 MB  [N][K]
  bf16_t* wkT    = (bf16_t*)(ws + off); off += (size_t)HS_ * (KV_ * DH_) * 2; // 0.5 MB
  bf16_t* wvT    = (bf16_t*)(ws + off); off += (size_t)HS_ * (KV_ * DH_) * 2; // 0.5 MB
  bf16_t* woT    = (bf16_t*)(ws + off); off += (size_t)(H_ * DH_) * HS_ * 2;  // 2 MB
  bf16_t* Qbuf   = (bf16_t*)(ws + off); off += (size_t)MTOT * (H_ * DH_) * 2; // 8 MB
  bf16_t* Kbuf   = (bf16_t*)(ws + off); off += (size_t)MTOT * (KV_ * DH_) * 2;// 2 MB
  bf16_t* Vtbuf  = (bf16_t*)(ws + off); off += (size_t)MTOT * (KV_ * DH_) * 2;// 2 MB [b][kv][dh][S]
  bf16_t* AObuf  = (bf16_t*)(ws + off); off += (size_t)MTOT * (H_ * DH_) * 2; // 8 MB

  // 1) conversions: hidden linear, weights transpose-converted to [N][K]
  cvt_bf16_kernel<<<(MTOT * HS_) / 1024, 256, 0, stream>>>(hidden, hid_bf, MTOT * HS_);
  cvt_t_bf16_kernel<<<dim3((H_ * DH_) / 32, HS_ / 32), 256, 0, stream>>>(wq, wqT, HS_, H_ * DH_);
  cvt_t_bf16_kernel<<<dim3((KV_ * DH_) / 32, HS_ / 32), 256, 0, stream>>>(wk, wkT, HS_, KV_ * DH_);
  cvt_t_bf16_kernel<<<dim3((KV_ * DH_) / 32, HS_ / 32), 256, 0, stream>>>(wv, wvT, HS_, KV_ * DH_);
  cvt_t_bf16_kernel<<<dim3(HS_ / 32, (H_ * DH_) / 32), 256, 0, stream>>>(wo, woT, H_ * DH_, HS_);

  // 2) projections (Q/K fused RMS norm; V scattered to transposed layout)
  gemm_bf16_kernel<<<dim3((H_ * DH_) / BN, MTOT / BM), 256, 0, stream>>>(
      hid_bf, wqT, nullptr, Qbuf, qnw, 0, MTOT, H_ * DH_, HS_);
  gemm_bf16_kernel<<<dim3((KV_ * DH_) / BN, MTOT / BM), 256, 0, stream>>>(
      hid_bf, wkT, nullptr, Kbuf, knw, 0, MTOT, KV_ * DH_, HS_);
  gemm_bf16_kernel<<<dim3((KV_ * DH_) / BN, MTOT / BM), 256, 0, stream>>>(
      hid_bf, wvT, nullptr, Vtbuf, nullptr, 1, MTOT, KV_ * DH_, HS_);

  // 3) flash attention
  attn_kernel<<<dim3(S_ / 64, H_, B_), 256, 0, stream>>>(Qbuf, Kbuf, Vtbuf, mask, AObuf);

  // 4) output projection -> fp32 d_out
  gemm_bf16_kernel<<<dim3(HS_ / BN, MTOT / BM), 256, 0, stream>>>(
      AObuf, woT, out, nullptr, nullptr, 0, MTOT, HS_, H_ * DH_);
}
